// SABLRetinaHeadWraper_1202590843783
// MI455X (gfx1250) — compile-verified, matrix operand order checked by reading, not run on hardware
//
#include <hip/hip_runtime.h>
#include <hip/hip_bf16.h>

// ---------------------------------------------------------------------------
// SABL RetinaHead post-processing for MI455X (gfx1250, wave32).
//
//  * 116 MB inputs @ 23.3 TB/s -> ~5us memory floor; workload is
//    selection/sort/NMS bound. Big tensors are read exactly once (K1),
//    survivors re-gathered later (L2-resident, 192MB).
//  * K1 was VMEM-issue bound (17M b32 loads > HBM floor): levels 0/1
//    (HW%4==0, 94% of data) use float4 channel loads (4x fewer VMEM
//    instructions); odd-sized levels 2-4 keep the scalar path.
//  * sigmoid is monotonic -> rank by max-over-class of raw logits.
//  * top-k via LDS bitonic sorts (<=64KB/WG, hierarchical chunking),
//    index tie-break matching jax.lax.top_k.
//  * NMS per (class,batch) WG: suppression adjacency as a 256x256 bitmask;
//    the IoU union term area_i+area_j per 16x16 tile is a rank-2 outer
//    product computed with v_wmma_f32_16x16x32_f16 (areas prescaled by
//    1/64^2 for f16 range); intersection min/max stays in f32 VALU and
//    co-executes with the 16-bit WMMA (tracked as TRANS per ISA).
//  * gfx1250 async path: GLOBAL_LOAD_ASYNC_TO_LDS_B32/B128 (ASYNCcnt) for
//    the per-lane scattered box gather and the bulk LDS sort fills, with
//    explicit s_wait_asynccnt 0 before barriers. Signatures confirmed from
//    compiler diagnostics: (AS1 int* src, AS3 int* dst, imm, imm) for b32,
//    AS-qualified int __attribute__((vector_size(16)))* for b128.
//  * __builtin_prefetch -> global_prefetch_b8 on the strided channel walk.
// ---------------------------------------------------------------------------

typedef __attribute__((ext_vector_type(16))) _Float16 v16h;
typedef __attribute__((ext_vector_type(8)))  float    v8f;

#define NLEV 5
#define BATCH 16
#define NCLS 80
#define NTOT 13343          // sum of level HW
#define MLSTRIDE 13344      // padded per-batch stride for maxlogit
#define CAND_STRIDE 5120    // per-batch candidate stride (5000 used)
#define SCORE_THR 0.05f
#define IOU_THR 0.5f
#define NEG_INF (-3.4e38f)
#define BOX_SCL (1.0f / 64.0f)            // coordinate prescale for f16 areas
#define BOX_SCL2 (BOX_SCL * BOX_SCL)

// ---- gfx1250 async LDS copies (ASYNCcnt), probe-guarded -------------------
#if __has_builtin(__builtin_amdgcn_global_load_async_to_lds_b32) && \
    __has_builtin(__builtin_amdgcn_global_load_async_to_lds_b128) && \
    __has_builtin(__builtin_amdgcn_s_wait_asynccnt)
#define HAS_ASYNC_LDS 1
typedef int i32x4_vs __attribute__((vector_size(16)));
typedef __attribute__((address_space(1))) int      g_i32;   // global b32
typedef __attribute__((address_space(3))) int      l_i32;   // LDS b32
typedef __attribute__((address_space(1))) i32x4_vs g_i32x4; // global b128
typedef __attribute__((address_space(3))) i32x4_vs l_i32x4; // LDS b128
#define ASYNC_COPY_B32(dst, src)                                            \
  __builtin_amdgcn_global_load_async_to_lds_b32((g_i32*)(src),              \
                                                (l_i32*)(dst), 0, 0)
#define ASYNC_COPY_B128(dst, src)                                           \
  __builtin_amdgcn_global_load_async_to_lds_b128((g_i32x4*)(src),           \
                                                 (l_i32x4*)(dst), 0, 0)
#define ASYNC_WAIT() __builtin_amdgcn_s_wait_asynccnt(0)
#else
#define ASYNC_COPY_B32(dst, src)  do { *(dst) = *(src); } while (0)
#define ASYNC_COPY_B128(dst, src) do { *(dst) = *(src); } while (0)
#define ASYNC_WAIT()              do {} while (0)
#endif

struct Ptrs {
  const float* cls[NLEV];
  const float* bcp[NLEV];
  const float* brp[NLEV];
};

__device__ __forceinline__ void level_of(int g, int& l, int& loc, int& hw) {
  if (g < 10000)      { l = 0; loc = g;         hw = 10000; }
  else if (g < 12500) { l = 1; loc = g - 10000; hw = 2500;  }
  else if (g < 13125) { l = 2; loc = g - 12500; hw = 625;   }
  else if (g < 13294) { l = 3; loc = g - 13125; hw = 169;   }
  else                { l = 4; loc = g - 13294; hw = 49;    }
}

__device__ __forceinline__ float fsigmoid(float x) {
  return 1.0f / (1.0f + __expf(-x));
}

// "a ranks before b": higher score first, lower index on ties (lax.top_k).
__device__ __forceinline__ bool rank_before(float sa, int ia, float sb, int ib) {
  return (sa > sb) || ((sa == sb) && (ia < ib));
}

// In-LDS bitonic sort, best element at position 0. n = power of two.
__device__ void block_bitonic_sort(float* key, int* idx, int n) {
  const int tid = threadIdx.x;
  const int T = blockDim.x;
  for (int k = 2; k <= n; k <<= 1) {
    for (int j = k >> 1; j > 0; j >>= 1) {
      __syncthreads();
      for (int t = tid; t < n; t += T) {
        int p = t ^ j;
        if (p > t) {
          bool up = ((t & k) == 0);
          float sa = key[t], sb = key[p];
          int ia = idx[t], ib = idx[p];
          bool ab = rank_before(sa, ia, sb, ib);
          if (up != ab) {
            key[t] = sb; key[p] = sa;
            idx[t] = ib; idx[p] = ia;
          }
        }
      }
    }
  }
  __syncthreads();
}

// ---------------------------------------------------------------------------
// K1a: max over 80 class logits, levels 0-1 (HW%4==0) as float4 quads.
// ---------------------------------------------------------------------------
__global__ void __launch_bounds__(256)
k_maxlogit_v4(const float* c0, const float* c1, float* ml) {
  int i = blockIdx.x * blockDim.x + threadIdx.x;
  if (i >= BATCH * 3125) return;                // 2500 + 625 quads
  int b = i / 3125, q = i % 3125;
  const float* p; int hw, loc, off;
  if (q < 2500) { p = c0; hw = 10000; loc = q * 4;          off = 0;     }
  else          { p = c1; hw = 2500;  loc = (q - 2500) * 4; off = 10000; }
  p += (size_t)b * NCLS * hw + loc;             // 16B aligned (hw%4==0)
  float4 m = make_float4(NEG_INF, NEG_INF, NEG_INF, NEG_INF);
  for (int c = 0; c < NCLS; ++c) {
    if ((c & 7) == 0 && c + 8 < NCLS)
      __builtin_prefetch(p + (size_t)(c + 8) * hw, 0, 0);  // global_prefetch_b8
    const float4 v = *(const float4*)(p + (size_t)c * hw);
    m.x = fmaxf(m.x, v.x); m.y = fmaxf(m.y, v.y);
    m.z = fmaxf(m.z, v.z); m.w = fmaxf(m.w, v.w);
  }
  *(float4*)(ml + b * MLSTRIDE + off + loc) = m;
}

// K1b: scalar tail for levels 2-4 (odd HW: 625/169/49).
__global__ void __launch_bounds__(256)
k_maxlogit_tail(const float* c2, const float* c3, const float* c4, float* ml) {
  int i = blockIdx.x * blockDim.x + threadIdx.x;
  if (i >= BATCH * 843) return;                 // 625 + 169 + 49
  int b = i / 843, r = i % 843;
  const float* p; int hw, loc, off;
  if (r < 625)      { p = c2; hw = 625; loc = r;       off = 12500; }
  else if (r < 794) { p = c3; hw = 169; loc = r - 625; off = 13125; }
  else              { p = c4; hw = 49;  loc = r - 794; off = 13294; }
  p += (size_t)b * NCLS * hw + loc;
  float m = NEG_INF;
  for (int c = 0; c < NCLS; ++c) m = fmaxf(m, p[(size_t)c * hw]);
  ml[b * MLSTRIDE + off + loc] = m;
}

// ---------------------------------------------------------------------------
// K2a/K2b: level-0 (N=10000) top-1000, hierarchical (2 chunks of 5000).
// ---------------------------------------------------------------------------
__global__ void __launch_bounds__(1024)
k_topk_l0_chunk(const float* ml, float* tk, int* ti) {
  __shared__ float skey[8192];
  __shared__ int   sidx[8192];
  int b = blockIdx.x >> 1, ch = blockIdx.x & 1;
  const float* src = ml + b * MLSTRIDE + ch * 5000;
  for (int t = threadIdx.x; t < 8192; t += blockDim.x) {
    if (t < 5000) ASYNC_COPY_B32(&skey[t], src + t);
    else          skey[t] = NEG_INF;
    sidx[t] = ch * 5000 + t;
  }
  ASYNC_WAIT();
  block_bitonic_sort(skey, sidx, 8192);
  float* ok = tk + (b * 2 + ch) * 1024;
  int*   oi = ti + (b * 2 + ch) * 1024;
  for (int t = threadIdx.x; t < 1024; t += blockDim.x) {
    ok[t] = (t < 1000) ? skey[t] : NEG_INF;
    oi[t] = sidx[t];
  }
}

__global__ void __launch_bounds__(256)
k_topk_l0_merge(const float* tk, const int* ti, int* sel) {
  __shared__ float skey[2048];
  __shared__ int   sidx[2048];
  int b = blockIdx.x;
  for (int t = threadIdx.x; t < 2048; t += blockDim.x) {
    ASYNC_COPY_B32(&skey[t], tk + b * 2048 + t);
    ASYNC_COPY_B32(&sidx[t], ti + b * 2048 + t);
  }
  ASYNC_WAIT();
  block_bitonic_sort(skey, sidx, 2048);
  int* out = sel + (b * NLEV + 0) * 1024;
  for (int t = threadIdx.x; t < 1000; t += blockDim.x) out[t] = sidx[t];
}

// ---------------------------------------------------------------------------
// K2c: direct top-1000 for levels 1..4 (n2 <= 4096 -> <= 32KB LDS).
// For N < 1000 the tail holds padded indices (>= N), detected in K3.
// ---------------------------------------------------------------------------
__global__ void __launch_bounds__(256)
k_topk_level(const float* ml, int off, int N, int n2, int* sel, int lvl) {
  __shared__ float skey[4096];
  __shared__ int   sidx[4096];
  int b = blockIdx.x;
  const float* src = ml + b * MLSTRIDE + off;
  for (int t = threadIdx.x; t < n2; t += blockDim.x) {
    if (t < N) ASYNC_COPY_B32(&skey[t], src + t);
    else       skey[t] = NEG_INF;
    sidx[t] = t;
  }
  ASYNC_WAIT();
  block_bitonic_sort(skey, sidx, n2);
  int* out = sel + (b * NLEV + lvl) * 1024;
  for (int t = threadIdx.x; t < 1000; t += blockDim.x) out[t] = sidx[t];
}

// ---------------------------------------------------------------------------
// K3: bucket decode (softmax over 7 buckets, top-2, BucketingBBoxCoder) for
// the 1000 survivors per (batch, level). Anchors computed analytically.
// ---------------------------------------------------------------------------
__global__ void __launch_bounds__(256)
k_decode(Ptrs P, const float* ml, const int* sel,
         float4* cbox, float* cconf, float* cmax, int* cloc) {
  int i = blockIdx.x * blockDim.x + threadIdx.x;
  if (i >= BATCH * 5000) return;
  int b = i / 5000, r = i % 5000, l = r / 1000, t = r % 1000;
  const int HW[NLEV]  = {10000, 2500, 625, 169, 49};
  const int OFF[NLEV] = {0, 10000, 12500, 13125, 13294};
  const int WW[NLEV]  = {100, 50, 25, 13, 7};
  const int ST[NLEV]  = {8, 16, 32, 64, 128};
  int li = sel[(b * NLEV + l) * 1024 + t];
  int ci = b * CAND_STRIDE + l * 1000 + t;
  int hw = HW[l];
  if (li >= hw) {  // padded slot (levels with N < 1000): score 0, never kept
    cbox[ci] = make_float4(0.f, 0.f, 0.f, 0.f);
    cconf[ci] = 0.f; cmax[ci] = 0.f; cloc[ci] = -1;
    return;
  }
  int W = WW[l], stride = ST[l];
  int row = li / W, col = li % W;
  float cx = col * (float)stride, cy = row * (float)stride;
  float s = 4.0f * (float)stride;       // anchor side
  float half = 1.5f * s;                // 0.5 * (3*s) after bbox_rescale
  float bw = (3.0f * s) / 14.0f;        // bucket width (== bucket height)
  const float* bcp = P.bcp[l] + (size_t)b * 28 * hw + li;
  const float* brp = P.brp[l] + (size_t)b * 28 * hw + li;
  float pbase[4] = {cx - half, cx + half, cy - half, cy + half};
  float sgn[4]   = {1.f, -1.f, 1.f, -1.f};
  float edges[4];
  float conf = 0.f;
  for (int side = 0; side < 4; ++side) {
    float v[7];
    float m1 = NEG_INF; int l1 = 0;
    for (int k = 0; k < 7; ++k) {
      v[k] = bcp[(size_t)(side * 7 + k) * hw];
      if (v[k] > m1) { m1 = v[k]; l1 = k; }     // first max (tie -> low idx)
    }
    float m2 = NEG_INF; int l2 = 0;
    for (int k = 0; k < 7; ++k)
      if (k != l1 && v[k] > m2) { m2 = v[k]; l2 = k; }
    float sum = 0.f;
    for (int k = 0; k < 7; ++k) sum += __expf(v[k] - m1);
    float p1 = 1.0f / sum;
    float p2 = __expf(m2 - m1) / sum;
    int adj = (l2 - l1 == 1) || (l1 - l2 == 1);
    conf += p1 + (adj ? p2 : 0.f);
    float offv = brp[(size_t)(side * 7 + l1) * hw];
    float edge = pbase[side] + sgn[side] * ((float)l1 + 0.5f) * bw;
    edges[side] = fminf(fmaxf(edge - offv * bw, 0.f), 799.f);
  }
  conf *= 0.25f;
  cbox[ci] = make_float4(edges[0], edges[2], edges[1], edges[3]);
  cconf[ci] = conf;
  cmax[ci] = fsigmoid(ml[b * MLSTRIDE + OFF[l] + li]) * conf;
  cloc[ci] = OFF[l] + li;
}

// ---------------------------------------------------------------------------
// K4: per-batch top-1000 of the 5000 candidates (by conf*max-sigmoid).
// ---------------------------------------------------------------------------
__global__ void __launch_bounds__(1024)
k_topk_batch(const float4* cbox, const float* cconf, const float* cmax,
             const int* cloc, float4* fbox, float* fconf, int* floc) {
  __shared__ float skey[8192];
  __shared__ int   sidx[8192];
  int b = blockIdx.x;
  for (int t = threadIdx.x; t < 8192; t += blockDim.x) {
    if (t < 5000) ASYNC_COPY_B32(&skey[t], cmax + b * CAND_STRIDE + t);
    else          skey[t] = -1.f;
    sidx[t] = t;
  }
  ASYNC_WAIT();
  block_bitonic_sort(skey, sidx, 8192);
  for (int t = threadIdx.x; t < 1024; t += blockDim.x) {
    if (t < 1000) {
      int j = sidx[t];
      fbox[b * 1024 + t]  = cbox[b * CAND_STRIDE + j];
      fconf[b * 1024 + t] = cconf[b * CAND_STRIDE + j];
      floc[b * 1024 + t]  = cloc[b * CAND_STRIDE + j];
    } else {
      fbox[b * 1024 + t]  = make_float4(0.f, 0.f, 0.f, 0.f);
      fconf[b * 1024 + t] = 0.f;
      floc[b * 1024 + t]  = -1;
    }
  }
}

// ---------------------------------------------------------------------------
// K5: per-(class,batch) NMS. 256 threads, thread t owns sorted slot t.
//  1. class score = sigmoid(logit)*conf for 1000 candidates; bitonic top-256.
//  2. per-lane scattered box gather -> LDS via GLOBAL_LOAD_ASYNC_TO_LDS_B128.
//  3. adjacency (IoU > 0.5, col > row) as 256x256 bitmask; per 16x16 tile
//     the union term area_i+area_j comes from a WMMA rank-2 outer product.
//  4. greedy scan over the bitmask (1 barrier / step).
// ---------------------------------------------------------------------------
__global__ void __launch_bounds__(256)
k_nms(Ptrs P, const float4* fbox, const float* fconf, const int* floc,
      float* kept_s, float4* kept_box) {
  __shared__ float  skey[1024];
  __shared__ int    sidx[1024];
  __shared__ float4 sbox[256];
  __shared__ float  sarea[256];         // prescaled areas
  __shared__ unsigned adjT[256 * 8];    // column-major bitmask
  __shared__ int    keptL[256];
  const int c = blockIdx.x, b = blockIdx.y;
  const int tid = threadIdx.x;

  for (int t = tid; t < 1024; t += 256) {
    float sv = -1.f;
    if (t < 1000) {
      sv = 0.f;
      int g = floc[b * 1024 + t];
      if (g >= 0) {
        int l, loc, hw;
        level_of(g, l, loc, hw);
        float logit = P.cls[l][((size_t)b * NCLS + c) * hw + loc];
        sv = fsigmoid(logit) * fconf[b * 1024 + t];
      }
    }
    skey[t] = sv; sidx[t] = t;
  }
  block_bitonic_sort(skey, sidx, 1024);

  float sc = skey[tid];
  int ci = sidx[tid];                   // always < 1000 (real candidates)
  bool validt = sc > SCORE_THR;
  // Scattered per-lane gather: each lane its own global + LDS address.
  ASYNC_COPY_B128(&sbox[tid], &fbox[b * 1024 + ci]);
  ASYNC_WAIT();
  float4 bx = sbox[tid];
  float aw = fmaxf(bx.z - bx.x, 0.f) * BOX_SCL;
  float ah = fmaxf(bx.w - bx.y, 0.f) * BOX_SCL;
  sarea[tid] = aw * ah;                 // <= ~156, exact enough in f16
  for (int t = tid; t < 2048; t += 256) adjT[t] = 0u;
  __syncthreads();

  // --- WMMA-assisted adjacency over 16x16 tiles (upper triangle only) ---
  const int wave = tid >> 5, lane = tid & 31;
  for (int tile = wave; tile < 256; tile += 8) {     // uniform per wave
    int ti = tile >> 4, tj = tile & 15;
    if (tj < ti) continue;                           // all cols <= rows
    // A[16x32] f16: row M=lane (lanes 0-15 hold K=0,1 in elems 0,1).
    // B[32x16] f16: col N=lane (lanes 0-15 hold K=0,1 in elems 0,1).
    // D = area_row * 1^T + 1 * area_col^T.
    v16h A = {};
    v16h Bm = {};
    if (lane < 16) {
      A[0]  = (_Float16)sarea[ti * 16 + lane];
      A[1]  = (_Float16)1.0f;
      Bm[0] = (_Float16)1.0f;
      Bm[1] = (_Float16)sarea[tj * 16 + lane];
    }
    v8f Cz = {};
    v8f S = __builtin_amdgcn_wmma_f32_16x16x32_f16(
        false, A, false, Bm, (short)0, Cz, false, false);
    // C/D layout: lane holds N = lane&15, rows M = v + 8*(lane>>4).
    int r0  = ti * 16 + ((lane >> 4) << 3);
    int cix = tj * 16 + (lane & 15);
    float4 bc = sbox[cix];
#pragma unroll
    for (int v = 0; v < 8; ++v) {
      int rr = r0 + v;
      if (cix > rr) {
        float4 br = sbox[rr];
        float iw = fminf(br.z, bc.z) - fmaxf(br.x, bc.x);
        float ih = fminf(br.w, bc.w) - fmaxf(br.y, bc.y);
        float inter = fmaxf(iw, 0.f) * fmaxf(ih, 0.f) * BOX_SCL2;
        float uni = S[v] - inter;
        if (inter > IOU_THR * fmaxf(uni, 1e-6f * BOX_SCL2))
          atomicOr(&adjT[cix * 8 + (rr >> 5)], 1u << (rr & 31));
      }
    }
  }
  __syncthreads();

  unsigned colw[8];
#pragma unroll
  for (int w = 0; w < 8; ++w) colw[w] = adjT[tid * 8 + w];

  bool sup = false;
  int mykeep = 0;
  for (int i = 0; i < 256; ++i) {
    if (tid == i) { mykeep = (validt && !sup) ? 1 : 0; keptL[i] = mykeep; }
    __syncthreads();
    if (keptL[i] && ((colw[i >> 5] >> (i & 31)) & 1u)) sup = true;
  }
  size_t o = ((size_t)c * BATCH + b) * 256 + tid;   // [C][B][256]
  kept_s[o] = mykeep ? sc : 0.f;
  kept_box[o] = bx;
}

// ---------------------------------------------------------------------------
// K6a/K6b: final per-batch top-100 of 80*256 = 20480 kept scores
// (10 chunks of 2048 -> top-128 each -> merge 1280).
// ---------------------------------------------------------------------------
__global__ void __launch_bounds__(256)
k_final_chunk(const float* kept_s, float* tk, int* ti) {
  __shared__ float skey[2048];
  __shared__ int   sidx[2048];
  int b = blockIdx.x / 10, ch = blockIdx.x % 10;
  for (int t = threadIdx.x; t < 2048; t += blockDim.x) {
    int g = ch * 2048 + t;              // 0..20479 = c*256 + k
    int c = g >> 8, k = g & 255;
    ASYNC_COPY_B32(&skey[t], kept_s + ((size_t)c * BATCH + b) * 256 + k);
    sidx[t] = g;
  }
  ASYNC_WAIT();
  block_bitonic_sort(skey, sidx, 2048);
  float* ok = tk + (b * 10 + ch) * 128;
  int*   oi = ti + (b * 10 + ch) * 128;
  for (int t = threadIdx.x; t < 128; t += blockDim.x) { ok[t] = skey[t]; oi[t] = sidx[t]; }
}

__global__ void __launch_bounds__(256)
k_final(const float* tk, const int* ti, const float4* kept_box, float* d_out) {
  __shared__ float skey[2048];
  __shared__ int   sidx[2048];
  __shared__ int   cnt;
  int b = blockIdx.x;
  for (int t = threadIdx.x; t < 2048; t += blockDim.x) {
    if (t < 1280) {
      ASYNC_COPY_B32(&skey[t], tk + b * 1280 + t);
      ASYNC_COPY_B32(&sidx[t], ti + b * 1280 + t);
    } else {
      skey[t] = -1.f; sidx[t] = 0;
    }
  }
  ASYNC_WAIT();
  block_bitonic_sort(skey, sidx, 2048);
  if (threadIdx.x == 0) cnt = 0;
  __syncthreads();
  // Output tuple flat: num_detected[16] i32 | proposals[16,100,4] f32 |
  //                    top_s[16,100] f32   | cls_id[16,100] i32   (9616 elems)
  int*   out_nd   = (int*)d_out;
  float* out_prop = d_out + 16;
  float* out_ts   = d_out + 16 + 6400;
  int*   out_cls  = (int*)(d_out + 16 + 6400 + 1600);
  for (int t = threadIdx.x; t < 100; t += blockDim.x) {
    float s = skey[t];
    int g = sidx[t];                    // 1280 real entries -> t<100 is real
    int c = g >> 8, k = g & 255;
    float4 bxv = kept_box[((size_t)c * BATCH + b) * 256 + k];
    out_ts[b * 100 + t] = s;
    out_cls[b * 100 + t] = c;
    float* pp = out_prop + (size_t)(b * 100 + t) * 4;
    pp[0] = bxv.x; pp[1] = bxv.y; pp[2] = bxv.z; pp[3] = bxv.w;
    if (s > SCORE_THR) atomicAdd(&cnt, 1);
  }
  __syncthreads();
  if (threadIdx.x == 0) out_nd[b] = cnt;
}

// ---------------------------------------------------------------------------
// Host launcher. Workspace layout (float units); total 2,712,064 floats
// (~10.4 MB) — every region fully written before read.
// ---------------------------------------------------------------------------
extern "C" void kernel_launch(void* const* d_in, const int* in_sizes, int n_in,
                              void* d_out, int out_size, void* d_ws, size_t ws_size,
                              hipStream_t stream) {
  (void)in_sizes; (void)n_in; (void)out_size; (void)ws_size;
  Ptrs P;
  for (int l = 0; l < NLEV; ++l) {
    P.cls[l] = (const float*)d_in[3 * l + 0];
    P.bcp[l] = (const float*)d_in[3 * l + 1];
    P.brp[l] = (const float*)d_in[3 * l + 2];
  }
  float* ws = (float*)d_ws;
  float*  ml       = ws;                          // 213504
  int*    sel      = (int*)   (ws + 213504);      //  81920
  float4* cbox     = (float4*)(ws + 295424);      // 327680
  float*  cconf    =           ws + 623104;       //  81920
  float*  cmax     =           ws + 705024;       //  81920
  int*    cloc     = (int*)   (ws + 786944);      //  81920
  float4* fbox     = (float4*)(ws + 868864);      //  65536
  float*  fconf    =           ws + 934400;       //  16384
  int*    floc     = (int*)   (ws + 950784);      //  16384
  float*  kept_s   =           ws + 967168;       // 327680
  float4* kept_box = (float4*)(ws + 1294848);     // 1310720
  float*  tk0      =           ws + 2605568;      //  32768
  int*    ti0      = (int*)   (ws + 2638336);     //  32768
  float*  tk6      =           ws + 2671104;      //  20480
  int*    ti6      = (int*)   (ws + 2691584);     //  20480 -> end 2712064

  k_maxlogit_v4<<<(BATCH * 3125 + 255) / 256, 256, 0, stream>>>(
      P.cls[0], P.cls[1], ml);
  k_maxlogit_tail<<<(BATCH * 843 + 255) / 256, 256, 0, stream>>>(
      P.cls[2], P.cls[3], P.cls[4], ml);

  k_topk_l0_chunk<<<BATCH * 2, 1024, 0, stream>>>(ml, tk0, ti0);
  k_topk_l0_merge<<<BATCH, 256, 0, stream>>>(tk0, ti0, sel);
  k_topk_level<<<BATCH, 256, 0, stream>>>(ml, 10000, 2500, 4096, sel, 1);
  k_topk_level<<<BATCH, 256, 0, stream>>>(ml, 12500,  625, 1024, sel, 2);
  k_topk_level<<<BATCH, 256, 0, stream>>>(ml, 13125,  169, 1024, sel, 3);
  k_topk_level<<<BATCH, 256, 0, stream>>>(ml, 13294,   49, 1024, sel, 4);

  k_decode<<<(BATCH * 5000 + 255) / 256, 256, 0, stream>>>(
      P, ml, sel, cbox, cconf, cmax, cloc);

  k_topk_batch<<<BATCH, 1024, 0, stream>>>(cbox, cconf, cmax, cloc,
                                           fbox, fconf, floc);

  dim3 g5(NCLS, BATCH);
  k_nms<<<g5, 256, 0, stream>>>(P, fbox, fconf, floc, kept_s, kept_box);

  k_final_chunk<<<BATCH * 10, 256, 0, stream>>>(kept_s, tk6, ti6);
  k_final<<<BATCH, 256, 0, stream>>>(tk6, ti6, kept_box, (float*)d_out);
}
